// PagedGPTBigCodeHeadless_38740605010125
// MI455X (gfx1250) — matrix operand order repaired and checked by reading, not compiled, not run on hardware
//
#include <hip/hip_runtime.h>
#include <hip/hip_bf16.h>
#include <math.h>

// Model dims (fixed by the reference)
#define V_SZ  49157
#define D_SZ  2048
#define H_SZ  16
#define DH_SZ 128
#define L_SZ  4
#define FF_SZ 8192
#define B_SZ  4
#define S_SZ  512
#define M_SZ  (B_SZ * S_SZ)   // 2048 tokens
#define LN_EPS 1e-5f

typedef __attribute__((ext_vector_type(16))) __bf16 v16bf;
typedef __attribute__((ext_vector_type(8)))  float  v8f;

union Frag { v16bf v; unsigned u[8]; };

// ---- CDNA5 async global->LDS path (guarded; sync fallback keeps compiling) ----
#if defined(__has_builtin)
#if __has_builtin(__builtin_amdgcn_global_load_async_to_lds_b128)
#define USE_ASYNC 1
#else
#define USE_ASYNC 0
#endif
#else
#define USE_ASYNC 0
#endif

// GNU-style int4 vector: the async builtin takes v4i addrspace(1)* / addrspace(3)*
typedef int v4i_gnu __attribute__((vector_size(16)));
typedef __attribute__((address_space(1))) v4i_gnu glb_v4i;
typedef __attribute__((address_space(3))) v4i_gnu lds_v4i;

__device__ __forceinline__ void stage16(const void* g, void* l) {
#if USE_ASYNC
  __builtin_amdgcn_global_load_async_to_lds_b128(
      (glb_v4i*)g, (lds_v4i*)l, 0, 0);
#else
  *(uint4*)l = *(const uint4*)g;
#endif
}

__device__ __forceinline__ void wait_async_le4() {
#if USE_ASYNC
#if __has_builtin(__builtin_amdgcn_s_wait_asynccnt)
  __builtin_amdgcn_s_wait_asynccnt(4);
#else
  asm volatile("s_wait_asynccnt 0x4" ::: "memory");
#endif
#endif
}
__device__ __forceinline__ void wait_async_0() {
#if USE_ASYNC
#if __has_builtin(__builtin_amdgcn_s_wait_asynccnt)
  __builtin_amdgcn_s_wait_asynccnt(0);
#else
  asm volatile("s_wait_asynccnt 0x0" ::: "memory");
#endif
#endif
}

// ---------------------------------------------------------------------------
// pos_ids = clip(cumsum(x != 0) - 1, 0)
// ---------------------------------------------------------------------------
__global__ void posids_kernel(const int* __restrict__ x, int* __restrict__ pid) {
  int b = blockIdx.x;
  if (threadIdx.x == 0) {
    int c = 0;
    for (int s = 0; s < S_SZ; ++s) {
      if (x[b * S_SZ + s] != 0) ++c;
      int p = c - 1;
      pid[b * S_SZ + s] = p < 0 ? 0 : p;
    }
  }
}

// ---------------------------------------------------------------------------
// h[b,s,:] = tok_emb[x] + (x!=0 ? pos_emb[pid] : 0)   (fp32 residual stream)
// ---------------------------------------------------------------------------
__global__ __launch_bounds__(256) void embed_kernel(
    const int* __restrict__ x, const int* __restrict__ pid,
    const float* __restrict__ tok_emb, const float* __restrict__ pos_emb,
    float* __restrict__ h) {
  int s = blockIdx.x, b = blockIdx.y;
  int tok = x[b * S_SZ + s];
  int p   = pid[b * S_SZ + s];
  size_t ob = ((size_t)(b * S_SZ + s)) * D_SZ;
  for (int d = threadIdx.x; d < D_SZ; d += 256) {
    float v = tok_emb[(size_t)tok * D_SZ + d];
    if (tok != 0) v += pos_emb[(size_t)p * D_SZ + d];
    h[ob + d] = v;
  }
}

// ---------------------------------------------------------------------------
// Weight transpose + fp32->bf16:  in [K,N] row-major  ->  out [N,K] bf16
// ---------------------------------------------------------------------------
__global__ __launch_bounds__(256) void transpose_cvt_kernel(
    const float* __restrict__ in, __bf16* __restrict__ out, int K, int N) {
  __shared__ float t[32][33];
  int nb = blockIdx.x * 32, kb = blockIdx.y * 32;
  int tx = threadIdx.x & 31, ty = threadIdx.x >> 5;   // 32 x 8
  #pragma unroll
  for (int j = 0; j < 32; j += 8)
    t[ty + j][tx] = in[(size_t)(kb + ty + j) * N + (nb + tx)];
  __syncthreads();
  #pragma unroll
  for (int j = 0; j < 32; j += 8)
    out[(size_t)(nb + ty + j) * K + (kb + tx)] = (__bf16)t[tx][ty + j];
}

// ---------------------------------------------------------------------------
// LayerNorm over D=2048 (fp32 in, OT out); one block per row
// ---------------------------------------------------------------------------
template <typename OT>
__global__ __launch_bounds__(256) void ln_kernel(
    const float* __restrict__ in, OT* __restrict__ out,
    const float* __restrict__ g, const float* __restrict__ bta) {
  __shared__ float s1[256], s2[256];
  size_t row = blockIdx.x;
  const float* p = in + row * D_SZ;
  float a = 0.f, b2 = 0.f;
  for (int d = threadIdx.x; d < D_SZ; d += 256) {
    float v = p[d]; a += v; b2 += v * v;
  }
  s1[threadIdx.x] = a; s2[threadIdx.x] = b2;
  __syncthreads();
  for (int off = 128; off > 0; off >>= 1) {
    if (threadIdx.x < off) {
      s1[threadIdx.x] += s1[threadIdx.x + off];
      s2[threadIdx.x] += s2[threadIdx.x + off];
    }
    __syncthreads();
  }
  float m   = s1[0] * (1.f / D_SZ);
  float var = s2[0] * (1.f / D_SZ) - m * m;
  float rs  = rsqrtf(var + LN_EPS);
  OT* q = out + row * D_SZ;
  for (int d = threadIdx.x; d < D_SZ; d += 256)
    q[d] = (OT)((p[d] - m) * rs * g[d] + bta[d]);
}

// ---------------------------------------------------------------------------
// Causal softmax: fp32 scores [B,H,S,S] -> bf16 probs; one block per row.
// ---------------------------------------------------------------------------
__global__ __launch_bounds__(256) void softmax_kernel(
    const float* __restrict__ sc, __bf16* __restrict__ pr) {
  __shared__ float red[256];
  size_t row = blockIdx.x;            // flattened over B*H*S
  int s = (int)(row % S_SZ);
  const float* p = sc + row * (size_t)S_SZ;
  __bf16* o = pr + row * (size_t)S_SZ;
  int valid = s + 1;
  float mx = -1e30f;
  for (int c = threadIdx.x; c < valid; c += 256) mx = fmaxf(mx, p[c]);
  red[threadIdx.x] = mx; __syncthreads();
  for (int off = 128; off > 0; off >>= 1) {
    if (threadIdx.x < off) red[threadIdx.x] = fmaxf(red[threadIdx.x], red[threadIdx.x + off]);
    __syncthreads();
  }
  mx = red[0]; __syncthreads();
  float sum = 0.f;
  for (int c = threadIdx.x; c < valid; c += 256) sum += __expf(p[c] - mx);
  red[threadIdx.x] = sum; __syncthreads();
  for (int off = 128; off > 0; off >>= 1) {
    if (threadIdx.x < off) red[threadIdx.x] += red[threadIdx.x + off];
    __syncthreads();
  }
  float inv = 1.f / red[0];
  for (int c = threadIdx.x; c < S_SZ; c += 256)
    o[c] = (__bf16)((c < valid) ? __expf(p[c] - mx) * inv : 0.f);
}

// ---------------------------------------------------------------------------
// Batched bf16-WMMA GEMM:  C = epilogue(alpha * A @ Bt^T + bias [+ resid])
//   A  bf16 [M,K] row-major (lda)
//   Bt bf16 [N,K] row-major (ldb = K-stride) -- weights pre-transposed
//   C  bf16 (COUT=0) or fp32 (COUT=1), ldc; TRANSC stores v^T [B][DH][S]
//   Block tile 128x128, K-step 32, 8 waves (2x4 16x16x32 WMMA each),
//   double-buffered LDS fed by GLOBAL_LOAD_ASYNC_TO_LDS_B128 (ASYNCcnt).
// EPI: 0 = bias, 1 = bias + tanh-GELU, 2 = bias + residual add
// ---------------------------------------------------------------------------
template <int EPI, int COUT, bool TRANSC>
__global__ __launch_bounds__(256) void gemm_bf16_wmma(
    const __bf16* __restrict__ A, const __bf16* __restrict__ Bt,
    const float* __restrict__ bias, const float* __restrict__ resid,
    void* __restrict__ Cv,
    int K, int lda, int ldb, int ldc, float alpha, int batchH,
    long long sAb, long long sAh, long long sBb, long long sBh,
    long long sCb, long long sCh) {
  // rows padded to 40 bf16 (80B): every 16B chunk stays 16B-aligned
  __shared__ __attribute__((aligned(16))) __bf16 lsA[2][128][40];
  __shared__ __attribute__((aligned(16))) __bf16 lsB[2][128][40];

  const int tid  = threadIdx.x;
  const int lane = tid & 31;
  const int wave = tid >> 5;

  const int z  = blockIdx.z;
  const int zb = z / batchH;
  const int zh = z - zb * batchH;
  A  += (size_t)zb * sAb + (size_t)zh * sAh;
  Bt += (size_t)zb * sBb + (size_t)zh * sBh;
  const size_t coff = (size_t)zb * sCb + (size_t)zh * sCh;

  const int blockM = blockIdx.x * 128;
  const int blockN = blockIdx.y * 128;
  const int wm = (wave >> 1) * 32;
  const int wn = (wave & 1) * 64;
  const int lrow = lane & 15;
  const int lhi  = lane >> 4;

  v8f acc[2][4] = {};

  // stage one 128x32 A tile + 128x32 B tile: 2 x 16B chunks per thread each
  auto stage = [&](int k0, int bsel) {
    #pragma unroll
    for (int i = 0; i < 2; ++i) {
      int c  = tid + i * 256;        // 512 chunks
      int r  = c >> 2;
      int kc = (c & 3) * 8;          // bf16 elements
      stage16(A  + (size_t)(blockM + r) * lda + (k0 + kc), &lsA[bsel][r][kc]);
      stage16(Bt + (size_t)(blockN + r) * ldb + (k0 + kc), &lsB[bsel][r][kc]);
    }
  };

  const int nk = K >> 5;
  stage(0, 0);
  for (int t = 0; t < nk; ++t) {
    const int cur = t & 1;
    if (t + 1 < nk) {
      stage((t + 1) << 5, cur ^ 1);  // next tile flies behind the WMMAs
      wait_async_le4();              // the 4 oldest (tile t) are complete
    } else {
      wait_async_0();
    }
    __syncthreads();

    Frag af[2], bfr[4];
    #pragma unroll
    for (int mt = 0; mt < 2; ++mt) {
      int r  = wm + mt * 16 + lrow;
      int kb = lhi * 8;
      #pragma unroll
      for (int j = 0; j < 8; ++j) {
        int kk2 = ((j >> 2) * 16) + kb + ((j & 3) * 2);
        af[mt].u[j] = *reinterpret_cast<const unsigned*>(&lsA[cur][r][kk2]);
      }
    }
    #pragma unroll
    for (int nt = 0; nt < 4; ++nt) {
      int c  = wn + nt * 16 + lrow;
      int kb = lhi * 8;
      #pragma unroll
      for (int j = 0; j < 8; ++j)
        bfr[nt].u[j] = *reinterpret_cast<const unsigned*>(&lsB[cur][c][((j >> 2) * 16) + kb + ((j & 3) * 2)]);
    }
    #pragma unroll
    for (int mt = 0; mt < 2; ++mt)
      #pragma unroll
      for (int nt = 0; nt < 4; ++nt)
        acc[mt][nt] = __builtin_amdgcn_wmma_f32_16x16x32_bf16(
            false, af[mt].v, false, bfr[nt].v,
            (short)0, acc[mt][nt], false, false);
    __syncthreads();
  }

  // ---- epilogue: C/D layout -> VGPR r holds M = r + 8*(lane>=16), N = lane&15
  float*  Cf = (float*)Cv;
  __bf16* Cb = (__bf16*)Cv;
  #pragma unroll
  for (int mt = 0; mt < 2; ++mt) {
    #pragma unroll
    for (int nt = 0; nt < 4; ++nt) {
      int gn = blockN + wn + nt * 16 + lrow;
      float bval = bias ? bias[gn] : 0.f;
      #pragma unroll
      for (int r = 0; r < 8; ++r) {
        int gm = blockM + wm + mt * 16 + r + lhi * 8;
        float v = acc[mt][nt][r] * alpha + bval;
        if (EPI == 1) {
          float x3 = v * v * v;
          v = 0.5f * v * (1.f + tanhf(0.7978845608028654f * (v + 0.044715f * x3)));
        } else if (EPI == 2) {
          v += resid[coff + (size_t)gm * ldc + gn];
        }
        size_t ci;
        if (TRANSC)   // store v^T: [B][DH][S], gm = b*S+s, gn = d
          ci = coff + (size_t)(gm >> 9) * (DH_SZ * S_SZ) + (size_t)gn * S_SZ + (gm & (S_SZ - 1));
        else
          ci = coff + (size_t)gm * ldc + gn;
        if (COUT == 1) Cf[ci] = v; else Cb[ci] = (__bf16)v;
      }
    }
  }
}

// ---------------------------------------------------------------------------
// Host: full forward pass.
// d_in order (setup_inputs insertion order): x, tok_emb, pos_emb,
//   per layer {ln_g, ln_b, wq.w, wq.b, wk.w, wk.b, wv.w, wv.b, wo.w, wo.b,
//              ffln_g, ffln_b, w1.w, w1.b, w2.w, w2.b}, f_g, f_b
// ---------------------------------------------------------------------------
extern "C" void kernel_launch(void* const* d_in, const int* in_sizes, int n_in,
                              void* d_out, int out_size, void* d_ws, size_t ws_size,
                              hipStream_t stream) {
  (void)in_sizes; (void)n_in; (void)out_size; (void)ws_size;
  const int*   x       = (const int*)d_in[0];
  const float* tok_emb = (const float*)d_in[1];
  const float* pos_emb = (const float*)d_in[2];
  const float* f_g     = (const float*)d_in[3 + 16 * L_SZ + 0];
  const float* f_b     = (const float*)d_in[3 + 16 * L_SZ + 1];

  // ---- workspace layout (all region sizes are multiples of 256B) ----
  char* base = (char*)d_ws;
  const size_t nMD  = (size_t)M_SZ * D_SZ;                 // 4Mi
  const size_t nSC  = (size_t)B_SZ * H_SZ * S_SZ * S_SZ;   // 16Mi (== M*FF)
  float*  h      = (float*)base;                 base += nMD * 4;
  float*  scores = (float*)base;                 base += nSC * 4;
  __bf16* y      = (__bf16*)base;                base += nMD * 2;
  __bf16* q      = (__bf16*)base;                base += nMD * 2;
  __bf16* o      = (__bf16*)base;                base += nMD * 2;
  __bf16* kbuf   = (__bf16*)base;                base += (size_t)M_SZ * DH_SZ * 2;
  __bf16* vT     = (__bf16*)base;                base += (size_t)B_SZ * DH_SZ * S_SZ * 2;
  __bf16* bigbf  = (__bf16*)base;                base += nSC * 2;   // probs / ffn hidden
  __bf16* wqT    = (__bf16*)base;                base += (size_t)D_SZ * D_SZ * 2;
  __bf16* wkT    = (__bf16*)base;                base += (size_t)D_SZ * DH_SZ * 2;
  __bf16* wvT    = (__bf16*)base;                base += (size_t)D_SZ * DH_SZ * 2;
  __bf16* woT    = (__bf16*)base;                base += (size_t)D_SZ * D_SZ * 2;
  __bf16* w1T    = (__bf16*)base;                base += (size_t)D_SZ * FF_SZ * 2;
  __bf16* w2T    = (__bf16*)base;                base += (size_t)FF_SZ * D_SZ * 2;
  int*    pid    = (int*)base;

  posids_kernel<<<B_SZ, 32, 0, stream>>>(x, pid);
  embed_kernel<<<dim3(S_SZ, B_SZ), 256, 0, stream>>>(x, pid, tok_emb, pos_emb, h);

  const float iscale = 0.08838834764831845f;   // 1/sqrt(DH)

  for (int l = 0; l < L_SZ; ++l) {
    const int bidx = 3 + l * 16;
    const float* ln_g = (const float*)d_in[bidx + 0];
    const float* ln_b = (const float*)d_in[bidx + 1];
    const float* wq_w = (const float*)d_in[bidx + 2];
    const float* wq_b = (const float*)d_in[bidx + 3];
    const float* wk_w = (const float*)d_in[bidx + 4];
    const float* wk_b = (const float*)d_in[bidx + 5];
    const float* wv_w = (const float*)d_in[bidx + 6];
    const float* wv_b = (const float*)d_in[bidx + 7];
    const float* wo_w = (const float*)d_in[bidx + 8];
    const float* wo_b = (const float*)d_in[bidx + 9];
    const float* ffg  = (const float*)d_in[bidx + 10];
    const float* ffb  = (const float*)d_in[bidx + 11];
    const float* w1_w = (const float*)d_in[bidx + 12];
    const float* w1_b = (const float*)d_in[bidx + 13];
    const float* w2_w = (const float*)d_in[bidx + 14];
    const float* w2_b = (const float*)d_in[bidx + 15];

    // weights -> bf16, transposed to [N,K]
    transpose_cvt_kernel<<<dim3(D_SZ/32,  D_SZ/32),  256, 0, stream>>>(wq_w, wqT, D_SZ, D_SZ);
    transpose_cvt_kernel<<<dim3(DH_SZ/32, D_SZ/32),  256, 0, stream>>>(wk_w, wkT, D_SZ, DH_SZ);
    transpose_cvt_kernel<<<dim3(DH_SZ/32, D_SZ/32),  256, 0, stream>>>(wv_w, wvT, D_SZ, DH_SZ);
    transpose_cvt_kernel<<<dim3(D_SZ/32,  D_SZ/32),  256, 0, stream>>>(wo_w, woT, D_SZ, D_SZ);
    transpose_cvt_kernel<<<dim3(FF_SZ/32, D_SZ/32),  256, 0, stream>>>(w1_w, w1T, D_SZ, FF_SZ);
    transpose_cvt_kernel<<<dim3(D_SZ/32,  FF_SZ/32), 256, 0, stream>>>(w2_w, w2T, FF_SZ, D_SZ);

    // y = LN(h)  (bf16)
    ln_kernel<<<M_SZ, 256, 0, stream>>>(h, y, ln_g, ln_b);
    // q = y@wq + bq (bf16)
    gemm_bf16_wmma<0, 0, false><<<dim3(16, 16, 1), 256, 0, stream>>>(
        y, wqT, wq_b, nullptr, q, D_SZ, D_SZ, D_SZ, D_SZ, 1.f,
        1, 0, 0, 0, 0, 0, 0);
    // k = y@wk + bk (bf16 [M,DH])
    gemm_bf16_wmma<0, 0, false><<<dim3(16, 1, 1), 256, 0, stream>>>(
        y, wkT, wk_b, nullptr, kbuf, D_SZ, D_SZ, D_SZ, DH_SZ, 1.f,
        1, 0, 0, 0, 0, 0, 0);
    // v = y@wv + bv, stored transposed (bf16 [B,DH,S])
    gemm_bf16_wmma<0, 0, true><<<dim3(16, 1, 1), 256, 0, stream>>>(
        y, wvT, wv_b, nullptr, vT, D_SZ, D_SZ, D_SZ, 0, 1.f,
        1, 0, 0, 0, 0, 0, 0);
    // scores[b,h] = scale * q_bh @ k_b^T  (fp32, batched z = b*H+h)
    gemm_bf16_wmma<0, 1, false><<<dim3(4, 4, B_SZ * H_SZ), 256, 0, stream>>>(
        q, kbuf, nullptr, nullptr, scores, DH_SZ, D_SZ, DH_SZ, S_SZ, iscale,
        H_SZ,
        (long long)S_SZ * D_SZ, (long long)DH_SZ,
        (long long)S_SZ * DH_SZ, 0,
        (long long)H_SZ * S_SZ * S_SZ, (long long)S_SZ * S_SZ);
    // causal softmax: fp32 scores -> bf16 probs
    softmax_kernel<<<B_SZ * H_SZ * S_SZ, 256, 0, stream>>>(scores, bigbf);
    // o[b,:,h,:] = probs @ v  (B operand is v^T [DH,S])
    gemm_bf16_wmma<0, 0, false><<<dim3(4, 1, B_SZ * H_SZ), 256, 0, stream>>>(
        bigbf, vT, nullptr, nullptr, o, S_SZ, S_SZ, S_SZ, D_SZ, 1.f,
        H_SZ,
        (long long)H_SZ * S_SZ * S_SZ, (long long)S_SZ * S_SZ,
        (long long)DH_SZ * S_SZ, 0,
        (long long)S_SZ * D_SZ, (long long)DH_SZ);
    // h = h + o@wo + bo  (fp32 residual)
    gemm_bf16_wmma<2, 1, false><<<dim3(16, 16, 1), 256, 0, stream>>>(
        o, woT, wo_b, h, h, D_SZ, D_SZ, D_SZ, D_SZ, 1.f,
        1, 0, 0, 0, 0, 0, 0);
    // y = LN(h)
    ln_kernel<<<M_SZ, 256, 0, stream>>>(h, y, ffg, ffb);
    // ffh = gelu(y@w1 + b1)  (bf16 [M,FF])
    gemm_bf16_wmma<1, 0, false><<<dim3(16, 64, 1), 256, 0, stream>>>(
        y, w1T, w1_b, nullptr, bigbf, D_SZ, D_SZ, D_SZ, FF_SZ, 1.f,
        1, 0, 0, 0, 0, 0, 0);
    // h = h + ffh@w2 + b2  (fp32 residual)
    gemm_bf16_wmma<2, 1, false><<<dim3(16, 16, 1), 256, 0, stream>>>(
        bigbf, w2T, w2_b, h, h, FF_SZ, FF_SZ, FF_SZ, D_SZ, 1.f,
        1, 0, 0, 0, 0, 0, 0);
  }

  // out = LN(h)  (fp32)
  ln_kernel<<<M_SZ, 256, 0, stream>>>(h, (float*)d_out, f_g, f_b);
}
